// TokenChoiceMoE_29643864277428
// MI455X (gfx1250) — matrix operand order repaired and checked by reading, not compiled, not run on hardware
//
#include <hip/hip_runtime.h>
#include <cstdint>
#include <cstddef>

// ---------------------------------------------------------------------------
// MoE (token-choice, top-2 of 8 experts + 2 shared experts) for gfx1250.
// BF16 WMMA (v_wmma_f32_16x16x32_bf16), f32 accumulation.
// Weights pre-converted to bf16 and pre-transposed to [N][K] once per call.
// Tile feeds use GLOBAL_LOAD_ASYNC_TO_LDS (ASYNCcnt) with double-buffered LDS
// so async copies overlap WMMA compute.
// ---------------------------------------------------------------------------

typedef __attribute__((ext_vector_type(16))) __bf16       v16bf;
typedef __attribute__((ext_vector_type(8)))  float        v8f;
typedef __attribute__((ext_vector_type(4)))  unsigned int u32x4;
typedef __attribute__((ext_vector_type(4)))  float        f32x4;
typedef int v4i __attribute__((__vector_size__(16)));     // matches builtin param

#define T_TOK  4096
#define DMODEL 2048
#define DFF    8192
#define NEXP   8
#define SFF    4096

#if defined(__has_builtin)
#if __has_builtin(__builtin_amdgcn_global_load_async_to_lds_b128) && \
    __has_builtin(__builtin_amdgcn_s_wait_asynccnt)
#define USE_ASYNC_LDS 1
#endif
#endif

#if defined(USE_ASYNC_LDS)
#define WAIT_ASYNC(n) __builtin_amdgcn_s_wait_asynccnt(n)
#else
#define WAIT_ASYNC(n)
#endif

static __device__ __forceinline__ unsigned short f2bf(float f) {
  unsigned int u = __float_as_uint(f);
  u += 0x7fffu + ((u >> 16) & 1u);      // round-to-nearest-even to bf16
  return (unsigned short)(u >> 16);
}

struct FragBits { u32x4 lo, hi; };
union  FragU    { FragBits b; v16bf v; };

static __device__ __forceinline__ v16bf make_frag(u32x4 lo, u32x4 hi) {
  FragU u; u.b.lo = lo; u.b.hi = hi; return u.v;
}

static __device__ __forceinline__ v8f wmma_bf16(v16bf a, v16bf b, v8f c) {
  // (neg_a, A, neg_b, B, c_mod, C, reuse_a, reuse_b)
  return __builtin_amdgcn_wmma_f32_16x16x32_bf16(false, a, false, b,
                                                 (short)0, c, false, false);
}

// 16-byte global->LDS copy: async DMA (ASYNCcnt) when the toolchain has it.
static __device__ __forceinline__ void cp16(const unsigned short* g, unsigned short* l) {
#if defined(USE_ASYNC_LDS)
  __builtin_amdgcn_global_load_async_to_lds_b128(
      (__attribute__((address_space(1))) v4i*)g,
      (__attribute__((address_space(3))) v4i*)l, 0, 0);
#else
  *(u32x4*)l = *(const u32x4*)g;
#endif
}

// ---------------------------------------------------------------------------
// Routing / small kernels
// ---------------------------------------------------------------------------

__global__ void k_zero(int* counts, int* cursors, float* aux) {
  int i = threadIdx.x;
  if (i < NEXP) { counts[i] = 0; cursors[i] = 0; }
  if (i < NEXP + 1) aux[i] = 0.f;
}

__global__ __launch_bounds__(256)
void k_gate(const float* __restrict__ x, const float* __restrict__ wg,
            int* __restrict__ tk_idx, float* __restrict__ tk_w,
            int* __restrict__ counts, float* __restrict__ aux) {
  int t = blockIdx.x * 256 + threadIdx.x;
  float lg[NEXP];
#pragma unroll
  for (int e = 0; e < NEXP; ++e) lg[e] = 0.f;
  const float* xr = x + (size_t)t * DMODEL;
  for (int d = 0; d < DMODEL; ++d) {
    float xv = xr[d];
#pragma unroll
    for (int e = 0; e < NEXP; ++e) lg[e] += xv * wg[d * NEXP + e];
  }
  float mx = lg[0];
#pragma unroll
  for (int e = 1; e < NEXP; ++e) mx = fmaxf(mx, lg[e]);
  float p[NEXP]; float sum = 0.f;
#pragma unroll
  for (int e = 0; e < NEXP; ++e) { p[e] = __expf(lg[e] - mx); sum += p[e]; }
  float inv = 1.f / sum;
#pragma unroll
  for (int e = 0; e < NEXP; ++e) p[e] *= inv;
  float lse = mx + logf(sum);
  atomicAdd(&aux[0], lse * lse);
#pragma unroll
  for (int e = 0; e < NEXP; ++e) atomicAdd(&aux[1 + e], p[e]);

  int b0 = 0;
#pragma unroll
  for (int e = 1; e < NEXP; ++e) if (p[e] > p[b0]) b0 = e;
  int b1 = (b0 == 0) ? 1 : 0;
#pragma unroll
  for (int e = 0; e < NEXP; ++e) if (e != b0 && p[e] > p[b1]) b1 = e;

  tk_idx[t * 2]     = b0;  tk_w[t * 2]     = p[b0];
  tk_idx[t * 2 + 1] = b1;  tk_w[t * 2 + 1] = p[b1];
  atomicAdd(&counts[b0], 1);
  atomicAdd(&counts[b1], 1);
}

__global__ void k_scan(const int* __restrict__ counts, const float* __restrict__ aux,
                       int* __restrict__ offsets, float* __restrict__ tail) {
  int off = 0; float fp = 0.f;
  for (int e = 0; e < NEXP; ++e) {
    offsets[e] = off; off += counts[e];
    fp += ((float)counts[e] / (float)T_TOK) * (aux[1 + e] / (float)T_TOK);
  }
  offsets[NEXP] = off;
  float auxl = 0.01f * (float)NEXP * fp;          // AUX_COEF * E * sum(f*P)
  float zl   = 0.001f * aux[0] / (float)T_TOK;    // Z_COEF * mean(lse^2)
  tail[0] = auxl + zl;
}

__global__ void k_slots(const int* __restrict__ tk_idx, const int* __restrict__ offsets,
                        int* __restrict__ cursors, int* __restrict__ slot_of,
                        int* __restrict__ slotTok) {
  int id = blockIdx.x * 256 + threadIdx.x;        // id in [0, 2*T)
  int e = tk_idx[id];
  int pos = atomicAdd(&cursors[e], 1);
  int slot = offsets[e] + pos;
  slot_of[id] = slot;
  slotTok[slot] = id >> 1;
}

__global__ void k_gather(const float* __restrict__ x, const int* __restrict__ slotTok,
                         unsigned short* __restrict__ Xb, unsigned short* __restrict__ Xg) {
  int b = blockIdx.x;
  if (b < T_TOK) {
    const float* src = x + (size_t)b * DMODEL;
    unsigned short* dst = Xb + (size_t)b * DMODEL;
    for (int i = threadIdx.x; i < DMODEL; i += 256) dst[i] = f2bf(src[i]);
  } else {
    int s = b - T_TOK;
    int t = slotTok[s];
    const float* src = x + (size_t)t * DMODEL;
    unsigned short* dst = Xg + (size_t)s * DMODEL;
    for (int i = threadIdx.x; i < DMODEL; i += 256) dst[i] = f2bf(src[i]);
  }
}

// ---------------------------------------------------------------------------
// Weight pre-pass: W[K][N] f32 -> WT[N][K] bf16 (LDS-tiled, coalesced both ways)
// ---------------------------------------------------------------------------

__global__ __launch_bounds__(256)
void k_transpose(const float* __restrict__ src, unsigned short* __restrict__ dst,
                 int K, int N, long mstride) {
  __shared__ __align__(16) unsigned short tile[64][72];   // 144B rows, 16B aligned
  const float* S = src + (size_t)blockIdx.z * mstride;
  unsigned short* D = dst + (size_t)blockIdx.z * mstride;
  const int nb = blockIdx.x * 64, kb = blockIdx.y * 64;
  const int tid = threadIdx.x;

  { // read 64(k) x 64(n) f32 tile, convert, store transposed into LDS
    int kk = tid >> 2, q = (tid & 3) * 16;
    const float* sp = S + (size_t)(kb + kk) * N + nb + q;
    f32x4 v0 = ((const f32x4*)sp)[0], v1 = ((const f32x4*)sp)[1];
    f32x4 v2 = ((const f32x4*)sp)[2], v3 = ((const f32x4*)sp)[3];
#pragma unroll
    for (int j = 0; j < 4; ++j) {
      tile[q + j][kk]      = f2bf(v0[j]);
      tile[q + 4 + j][kk]  = f2bf(v1[j]);
      tile[q + 8 + j][kk]  = f2bf(v2[j]);
      tile[q + 12 + j][kk] = f2bf(v3[j]);
    }
  }
  __syncthreads();
  { // write 64(n) rows of 64(k) bf16, coalesced
    int nn = tid >> 2, c = (tid & 3) * 16;
    unsigned short* dp = D + (size_t)(nb + nn) * K + kb + c;
    ((u32x4*)dp)[0] = *(const u32x4*)&tile[nn][c];
    ((u32x4*)dp)[1] = *(const u32x4*)&tile[nn][c + 8];
  }
}

// ---------------------------------------------------------------------------
// GEMM1: H = silu(A@W1) * (A@W2). A bf16 [M][K], W*T bf16 [N][K], H bf16 [M][ldh].
// Block tile 128x64, 8 waves (4Mx2N), wave tile 32x32, kchunk 32,
// double-buffered LDS with async copies overlapping WMMA.
// ---------------------------------------------------------------------------

__global__ __launch_bounds__(256)
void k_gemm1(const unsigned short* __restrict__ A, int K,
             const unsigned short* __restrict__ W1T, const unsigned short* __restrict__ W2T,
             long wstride, unsigned short* __restrict__ Hout, int ldh,
             const int* __restrict__ counts, const int* __restrict__ offsets) {
  __shared__ __align__(16) unsigned short As [2][128][40];
  __shared__ __align__(16) unsigned short Bs1[2][64][40];
  __shared__ __align__(16) unsigned short Bs2[2][64][40];

  const int tid = threadIdx.x;
  const int e = blockIdx.z;
  int valid = 128;
  int rowOffset = blockIdx.y * 128;
  if (counts) {                     // routed expert path, dynamic row count
    int c = counts[e];
    if (rowOffset >= c) return;     // uniform exit: EXEC stays all-ones below
    valid = c - rowOffset; if (valid > 128) valid = 128;
    rowOffset += offsets[e];
  }
  const long wOff = (long)e * wstride;
  const int nBase = blockIdx.x * 64;

  const int lane = tid & 31;
  const int waveId = tid >> 5;
  const int wm = (waveId & 3) * 32;   // 4 waves along M
  const int wn = (waveId >> 2) * 32;  // 2 waves along N
  const int hiL = lane >> 4;
  const int ml  = lane & 15;

  // Per-wave batch = 4 async copies (2 A + 2 B); row clamped so count is fixed.
  auto issue_tiles = [&](int kb, int b) {
    {
      int r = tid >> 1, hf = tid & 1;
      int rs = (r < valid) ? r : (valid - 1);       // clamp: fixed batch size
      const unsigned short* src = A + (size_t)(rowOffset + rs) * K + kb + hf * 16;
      unsigned short* dst = &As[b][r][hf * 16];
      cp16(src, dst);
      cp16(src + 8, dst + 8);
    }
    {
      int n = tid >> 2, c = (tid & 3) * 8;
      const unsigned short* s1 = W1T + wOff + (size_t)(nBase + n) * K + kb + c;
      const unsigned short* s2 = W2T + wOff + (size_t)(nBase + n) * K + kb + c;
      cp16(s1, &Bs1[b][n][c]);
      cp16(s2, &Bs2[b][n][c]);
      if (kb + 32 < K) {            // global_prefetch_b8 of next K strip
        __builtin_prefetch(s1 + 32, 0, 0);
        __builtin_prefetch(s2 + 32, 0, 0);
      }
    }
  };

  v8f acc1[2][2] = {};
  v8f acc2[2][2] = {};

  issue_tiles(0, 0);
  int cur = 0;
  for (int kb = 0; kb < K; kb += 32, cur ^= 1) {
    if (kb + 32 < K) {              // stage next chunk; overlaps compute below
      issue_tiles(kb + 32, cur ^ 1);
      WAIT_ASYNC(4);                // in-order: older batch (cur) fully landed
    } else {
      WAIT_ASYNC(0);
    }
    __syncthreads();

    v16bf af[2];
#pragma unroll
    for (int s = 0; s < 2; ++s) {
      int row = wm + s * 16 + ml;   // A frag: K groups {0-7,16-23} lo / {8-15,24-31} hi
      af[s] = make_frag(*(const u32x4*)&As[cur][row][hiL * 8],
                        *(const u32x4*)&As[cur][row][16 + hiL * 8]);
    }
#pragma unroll
    for (int u = 0; u < 2; ++u) {
      int n = wn + u * 16 + ml;     // B frag: K groups {0-7,16-23} lo / {8-15,24-31} hi
      v16bf bf1 = make_frag(*(const u32x4*)&Bs1[cur][n][hiL * 8],
                            *(const u32x4*)&Bs1[cur][n][16 + hiL * 8]);
      v16bf bf2 = make_frag(*(const u32x4*)&Bs2[cur][n][hiL * 8],
                            *(const u32x4*)&Bs2[cur][n][16 + hiL * 8]);
#pragma unroll
      for (int s = 0; s < 2; ++s) {
        acc1[s][u] = wmma_bf16(af[s], bf1, acc1[s][u]);
        acc2[s][u] = wmma_bf16(af[s], bf2, acc2[s][u]);
      }
    }
    __syncthreads();                // buf[cur] reads done -> reusable
  }

  // SwiGLU epilogue; C/D layout: lane%16 = N, vgpr v = M row (hi lanes +8)
#pragma unroll
  for (int s = 0; s < 2; ++s)
#pragma unroll
    for (int u = 0; u < 2; ++u)
#pragma unroll
      for (int v = 0; v < 8; ++v) {
        int m = wm + s * 16 + hiL * 8 + v;
        if (m < valid) {
          int col = nBase + wn + u * 16 + ml;
          float g = acc1[s][u][v];
          float hval = (g / (1.f + __expf(-g))) * acc2[s][u][v];
          Hout[(size_t)(rowOffset + m) * ldh + col] = f2bf(hval);
        }
      }
}

// ---------------------------------------------------------------------------
// GEMM2: Out = A@W3 (f32 out, optional accumulate). A bf16 [M][K], W3T bf16 [N][K].
// ---------------------------------------------------------------------------

__global__ __launch_bounds__(256)
void k_gemm2(const unsigned short* __restrict__ A, int K,
             const unsigned short* __restrict__ W3T, long wstride,
             float* __restrict__ Out, int ldo, int accumulate,
             const int* __restrict__ counts, const int* __restrict__ offsets) {
  __shared__ __align__(16) unsigned short As[2][128][40];
  __shared__ __align__(16) unsigned short Bs[2][64][40];

  const int tid = threadIdx.x;
  const int e = blockIdx.z;
  int valid = 128;
  int rowOffset = blockIdx.y * 128;
  if (counts) {
    int c = counts[e];
    if (rowOffset >= c) return;
    valid = c - rowOffset; if (valid > 128) valid = 128;
    rowOffset += offsets[e];
  }
  const long wOff = (long)e * wstride;
  const int nBase = blockIdx.x * 64;

  const int lane = tid & 31;
  const int waveId = tid >> 5;
  const int wm = (waveId & 3) * 32;
  const int wn = (waveId >> 2) * 32;
  const int hiL = lane >> 4;
  const int ml  = lane & 15;

  // Per-wave batch = 3 async copies (2 A + 1 B).
  auto issue_tiles = [&](int kb, int b) {
    {
      int r = tid >> 1, hf = tid & 1;
      int rs = (r < valid) ? r : (valid - 1);
      const unsigned short* src = A + (size_t)(rowOffset + rs) * K + kb + hf * 16;
      unsigned short* dst = &As[b][r][hf * 16];
      cp16(src, dst);
      cp16(src + 8, dst + 8);
    }
    {
      int n = tid >> 2, c = (tid & 3) * 8;
      const unsigned short* s3 = W3T + wOff + (size_t)(nBase + n) * K + kb + c;
      cp16(s3, &Bs[b][n][c]);
      if (kb + 32 < K) __builtin_prefetch(s3 + 32, 0, 0);
    }
  };

  v8f acc[2][2] = {};

  issue_tiles(0, 0);
  int cur = 0;
  for (int kb = 0; kb < K; kb += 32, cur ^= 1) {
    if (kb + 32 < K) {
      issue_tiles(kb + 32, cur ^ 1);
      WAIT_ASYNC(3);
    } else {
      WAIT_ASYNC(0);
    }
    __syncthreads();

    v16bf af[2];
#pragma unroll
    for (int s = 0; s < 2; ++s) {
      int row = wm + s * 16 + ml;
      af[s] = make_frag(*(const u32x4*)&As[cur][row][hiL * 8],
                        *(const u32x4*)&As[cur][row][16 + hiL * 8]);
    }
#pragma unroll
    for (int u = 0; u < 2; ++u) {
      int n = wn + u * 16 + ml;
      v16bf bfv = make_frag(*(const u32x4*)&Bs[cur][n][hiL * 8],
                            *(const u32x4*)&Bs[cur][n][16 + hiL * 8]);
#pragma unroll
      for (int s = 0; s < 2; ++s) acc[s][u] = wmma_bf16(af[s], bfv, acc[s][u]);
    }
    __syncthreads();
  }

#pragma unroll
  for (int s = 0; s < 2; ++s)
#pragma unroll
    for (int u = 0; u < 2; ++u)
#pragma unroll
      for (int v = 0; v < 8; ++v) {
        int m = wm + s * 16 + hiL * 8 + v;
        if (m < valid) {
          int col = nBase + wn + u * 16 + ml;
          size_t o = (size_t)(rowOffset + m) * ldo + col;
          float val = acc[s][u][v];
          Out[o] = accumulate ? (Out[o] + val) : val;
        }
      }
}

// ---------------------------------------------------------------------------
// Final combine: out[t] = SA[t] + w0*Y[slot0] + w1*Y[slot1]
// ---------------------------------------------------------------------------

__global__ void k_combine(const float* __restrict__ SA, const float* __restrict__ Y,
                          const int* __restrict__ slot_of, const float* __restrict__ tk_w,
                          float* __restrict__ out) {
  int t = blockIdx.x;
  int s0 = slot_of[t * 2], s1 = slot_of[t * 2 + 1];
  float w0 = tk_w[t * 2], w1 = tk_w[t * 2 + 1];
  for (int i = threadIdx.x; i < DMODEL; i += 256)
    out[(size_t)t * DMODEL + i] = SA[(size_t)t * DMODEL + i]
        + w0 * Y[(size_t)s0 * DMODEL + i]
        + w1 * Y[(size_t)s1 * DMODEL + i];
}

// ---------------------------------------------------------------------------
// Launch
// ---------------------------------------------------------------------------

extern "C" void kernel_launch(void* const* d_in, const int* in_sizes, int n_in,
                              void* d_out, int out_size, void* d_ws, size_t ws_size,
                              hipStream_t stream) {
  (void)in_sizes; (void)n_in; (void)out_size; (void)ws_size;
  const float* x   = (const float*)d_in[0];
  const float* wg  = (const float*)d_in[1];
  const float* ew1 = (const float*)d_in[2];
  const float* ew2 = (const float*)d_in[3];
  const float* ew3 = (const float*)d_in[4];
  const float* sw1 = (const float*)d_in[5];
  const float* sw2 = (const float*)d_in[6];
  const float* sw3 = (const float*)d_in[7];
  float* out = (float*)d_out;

  unsigned char* ws8 = (unsigned char*)d_ws;
  size_t off = 0;
  unsigned short* H  = (unsigned short*)(ws8 + off); off += (size_t)8192 * 8192 * 2;   // 128MB
  float*          Y  = (float*)(ws8 + off);          off += (size_t)8192 * 2048 * 4;   // 64MB
  float*          SA = (float*)(ws8 + off);          off += (size_t)4096 * 2048 * 4;   // 32MB
  unsigned short* Xg = (unsigned short*)(ws8 + off); off += (size_t)8192 * 2048 * 2;   // 32MB
  unsigned short* Xb = (unsigned short*)(ws8 + off); off += (size_t)4096 * 2048 * 2;   // 16MB
  unsigned char*  sm = ws8 + off;                    off += 262144;
  unsigned short* ew1T = (unsigned short*)(ws8 + off); off += (size_t)NEXP * DMODEL * DFF * 2;
  unsigned short* ew2T = (unsigned short*)(ws8 + off); off += (size_t)NEXP * DMODEL * DFF * 2;
  unsigned short* ew3T = (unsigned short*)(ws8 + off); off += (size_t)NEXP * DFF * DMODEL * 2;
  unsigned short* sw1T = (unsigned short*)(ws8 + off); off += (size_t)2 * DMODEL * SFF * 2;
  unsigned short* sw2T = (unsigned short*)(ws8 + off); off += (size_t)2 * DMODEL * SFF * 2;
  unsigned short* sw3T = (unsigned short*)(ws8 + off); off += (size_t)2 * SFF * DMODEL * 2;

  int*   tk_idx  = (int*)sm;
  float* tk_w    = (float*)(sm + 32768);
  int*   slot_of = (int*)(sm + 65536);
  int*   slotTok = (int*)(sm + 98304);
  int*   counts  = (int*)(sm + 131072);
  int*   cursors = (int*)(sm + 131072 + 64);
  int*   offsets = (int*)(sm + 131072 + 128);
  float* aux     = (float*)(sm + 131072 + 192);

  // --- weight pre-pass: f32 [K][N] -> bf16 [N][K] ---
  k_transpose<<<dim3(DFF / 64, DMODEL / 64, NEXP), 256, 0, stream>>>(
      ew1, ew1T, DMODEL, DFF, (long)DMODEL * DFF);
  k_transpose<<<dim3(DFF / 64, DMODEL / 64, NEXP), 256, 0, stream>>>(
      ew2, ew2T, DMODEL, DFF, (long)DMODEL * DFF);
  k_transpose<<<dim3(DMODEL / 64, DFF / 64, NEXP), 256, 0, stream>>>(
      ew3, ew3T, DFF, DMODEL, (long)DFF * DMODEL);
  k_transpose<<<dim3(SFF / 64, DMODEL / 64, 2), 256, 0, stream>>>(
      sw1, sw1T, DMODEL, SFF, (long)DMODEL * SFF);
  k_transpose<<<dim3(SFF / 64, DMODEL / 64, 2), 256, 0, stream>>>(
      sw2, sw2T, DMODEL, SFF, (long)DMODEL * SFF);
  k_transpose<<<dim3(DMODEL / 64, SFF / 64, 2), 256, 0, stream>>>(
      sw3, sw3T, SFF, DMODEL, (long)SFF * DMODEL);

  // --- routing ---
  k_zero<<<1, 32, 0, stream>>>(counts, cursors, aux);
  k_gate<<<T_TOK / 256, 256, 0, stream>>>(x, wg, tk_idx, tk_w, counts, aux);
  k_scan<<<1, 1, 0, stream>>>(counts, aux, offsets, out + (size_t)T_TOK * DMODEL);
  k_slots<<<(2 * T_TOK) / 256, 256, 0, stream>>>(tk_idx, offsets, cursors, slot_of, slotTok);
  k_gather<<<T_TOK + 2 * T_TOK, 256, 0, stream>>>(x, slotTok, Xb, Xg);

  // --- shared experts (H buffer reused, stream-ordered) ---
  for (int i = 0; i < 2; ++i) {
    k_gemm1<<<dim3(SFF / 64, T_TOK / 128, 1), 256, 0, stream>>>(
        Xb, DMODEL, sw1T + (size_t)i * DMODEL * SFF, sw2T + (size_t)i * DMODEL * SFF,
        0L, H, SFF, nullptr, nullptr);
    k_gemm2<<<dim3(DMODEL / 64, T_TOK / 128, 1), 256, 0, stream>>>(
        H, SFF, sw3T + (size_t)i * SFF * DMODEL, 0L, SA, DMODEL, i, nullptr, nullptr);
  }

  // --- routed experts (grouped GEMM over slot ranges) ---
  k_gemm1<<<dim3(DFF / 64, 32, NEXP), 256, 0, stream>>>(
      Xg, DMODEL, ew1T, ew2T, (long)DMODEL * DFF, H, DFF, counts, offsets);
  k_gemm2<<<dim3(DMODEL / 64, 32, NEXP), 256, 0, stream>>>(
      H, DFF, ew3T, (long)DFF * DMODEL, Y, DMODEL, 0, counts, offsets);

  // --- combine ---
  k_combine<<<T_TOK, 256, 0, stream>>>(SA, Y, slot_of, tk_w, out);
}
